// GenFlow_unit_78426102825581
// MI455X (gfx1250) — compile-verified, matrix operand order checked by reading, not run on hardware
//
#include <hip/hip_runtime.h>

typedef __attribute__((ext_vector_type(16))) _Float16 v16h;
typedef __attribute__((ext_vector_type(8)))  float    v8f;

#define DEVFN static __device__ __forceinline__

static constexpr int BB = 2;
static constexpr int NN = 8192;

DEVFN int lane_id() { return (int)(threadIdx.x & 31); }

// ------------------- WMMA 16x16x32 f16->f32 fragment helpers -------------------
// A fragment (16x32, 16-bit): lanes 0-15 -> M=lane, lanes 16-31 -> same M with K+8.
// VGPR j holds K = 2j + (j>=4 ? 8:0) (+8 for upper half-lanes), two K per dword.
// Per lane this is two contiguous 8-half chunks -> compiler emits b128 loads.
DEVFN v16h load_a_lds(const _Float16* lds, int row0, int k0, int stride) {
  int lane = lane_id(); int hi = (lane >> 4) & 1; int m = lane & 15;
  const _Float16* p = lds + (size_t)(row0 + m) * stride + k0 + (hi ? 8 : 0);
  v16h a;
#pragma unroll
  for (int j = 0; j < 8; ++j) {
    int base = 2 * j + (j >= 4 ? 8 : 0);
    a[2 * j]     = p[base];
    a[2 * j + 1] = p[base + 1];
  }
  return a;
}

// B fragment (32x16): B[k][n] = Wp[n][k], Wp pre-packed f16, zero-padded to KP.
// Same contiguous-chunk pattern as A -> two global_load_b128 per lane, no branches.
DEVFN v16h load_b_packed(const _Float16* Wp, int n0, int k0, int KP) {
  int lane = lane_id(); int hi = (lane >> 4) & 1; int n = n0 + (lane & 15);
  const _Float16* p = Wp + (size_t)n * KP + k0 + (hi ? 8 : 0);
  v16h b;
#pragma unroll
  for (int j = 0; j < 8; ++j) {
    int base = 2 * j + (j >= 4 ? 8 : 0);
    b[2 * j]     = p[base];
    b[2 * j + 1] = p[base + 1];
  }
  return b;
}

// C/D 16x16 f32: lanes 0-15 N=lane, vgpr r -> M=r; lanes 16-31 -> M=r+8.
DEVFN void store_d_relu(v8f d, _Float16* lds, int row0, int n0, int stride, const float* bias) {
  int lane = lane_id(); int hi = (lane >> 4) & 1; int n = n0 + (lane & 15);
  float bv = bias[n];
#pragma unroll
  for (int r = 0; r < 8; ++r) {
    float v = d[r] + bv;
    v = fmaxf(v, 0.f);
    lds[(size_t)(row0 + r + (hi ? 8 : 0)) * stride + n] = (_Float16)v;
  }
}

DEVFN v8f wmma_f16(v16h a, v16h b, v8f c) {
  return __builtin_amdgcn_wmma_f32_16x16x32_f16(false, a, false, b, (short)0, c, false, false);
}

// LDS GEMM: O[row,n] = relu(sum_k A[row,k] * Wp[n,k] + bias[n]); row tiles split over waves.
template <int ROWS, int KT, int COUT>
DEVFN void lds_gemm_relu(const _Float16* A, int strideA, const _Float16* Wp,
                         const float* bias, _Float16* O, int strideO, int nwaves) {
  constexpr int KP = KT * 32;
  int wave = (int)(threadIdx.x >> 5);
  for (int rt = wave; rt < ROWS / 16; rt += nwaves) {
    for (int nt = 0; nt < COUT / 16; ++nt) {
      v8f acc = {};
#pragma unroll
      for (int kt = 0; kt < KT; ++kt) {
        v16h a = load_a_lds(A, rt * 16, kt * 32, strideA);
        v16h b = load_b_packed(Wp, nt * 16, kt * 32, KP);
        acc = wmma_f16(a, b, acc);
      }
      store_d_relu(acc, O, rt * 16, nt * 16, strideO, bias);
    }
  }
}

// ------------------- weight packer: f32 [Cout][Cin] -> f16 [Cout][KP] zero-padded ---------------
__global__ void pack_w_kernel(const float* __restrict__ W, _Float16* __restrict__ out,
                              int Cout, int Cin, int KP) {
  int t = blockIdx.x * blockDim.x + threadIdx.x;
  if (t >= Cout * KP) return;
  int n = t / KP, k = t % KP;
  out[t] = (_Float16)((k < Cin) ? W[(size_t)n * Cin + k] : 0.f);
}

// ------------------- init: transpose pc/feat, f0 = relu(featT @ W^T + b) -------------------
__global__ void init_kernel(const float* __restrict__ pc, const float* __restrict__ feat,
                            const float* __restrict__ w, const float* __restrict__ bias,
                            float* __restrict__ xyz0, float* __restrict__ f0) {
  int t = blockIdx.x * blockDim.x + threadIdx.x;
  if (t >= BB * NN) return;
  int b = t / NN, n = t % NN;
  const float* P = pc + (size_t)b * 3 * NN;
  const float* F = feat + (size_t)b * 3 * NN;
  float fv[3];
#pragma unroll
  for (int c = 0; c < 3; ++c) {
    xyz0[((size_t)b * NN + n) * 3 + c] = P[(size_t)c * NN + n];
    fv[c] = F[(size_t)c * NN + n];
  }
  float* o = f0 + ((size_t)b * NN + n) * 32;
  for (int oc = 0; oc < 32; ++oc) {
    float a = bias[oc] + fv[0] * w[oc * 3] + fv[1] * w[oc * 3 + 1] + fv[2] * w[oc * 3 + 2];
    o[oc] = fmaxf(a, 0.f);
  }
}

// ------------------- farthest point sampling (one block per batch) -------------------
__global__ void fps_kernel(const float* __restrict__ xyz, int Nsrc, int M,
                           int* __restrict__ idx_out, float* __restrict__ newxyz) {
  __shared__ float sval[1024];
  __shared__ int sidx[1024];
  int b = blockIdx.x;
  const float* X = xyz + (size_t)b * Nsrc * 3;
  int tid = threadIdx.x;
  float px[8], py[8], pz[8], dm[8];
  float x0 = X[0], y0 = X[1], z0 = X[2];
  int cnt = 0;
  for (int i = tid; i < Nsrc && cnt < 8; i += 1024) {
    px[cnt] = X[i * 3]; py[cnt] = X[i * 3 + 1]; pz[cnt] = X[i * 3 + 2];
    float dx = px[cnt] - x0, dy = py[cnt] - y0, dz = pz[cnt] - z0;
    dm[cnt] = dx * dx + dy * dy + dz * dz;
    ++cnt;
  }
  if (tid == 0) {
    idx_out[(size_t)b * M] = 0;
    newxyz[(size_t)b * M * 3 + 0] = x0;
    newxyz[(size_t)b * M * 3 + 1] = y0;
    newxyz[(size_t)b * M * 3 + 2] = z0;
  }
  for (int s = 1; s < M; ++s) {
    float bv = -1.f; int bi = 0x7fffffff;
    for (int j = 0; j < cnt; ++j) {
      int i = tid + j * 1024;
      if (dm[j] > bv || (dm[j] == bv && i < bi)) { bv = dm[j]; bi = i; }
    }
    sval[tid] = bv; sidx[tid] = bi;
    __syncthreads();
    for (int off = 512; off > 0; off >>= 1) {
      if (tid < off) {
        float ov = sval[tid + off]; int oi = sidx[tid + off];
        if (ov > sval[tid] || (ov == sval[tid] && oi < sidx[tid])) { sval[tid] = ov; sidx[tid] = oi; }
      }
      __syncthreads();
    }
    int nxt = sidx[0];
    float nx = X[nxt * 3], ny = X[nxt * 3 + 1], nz = X[nxt * 3 + 2];
    if (tid == 0) {
      idx_out[(size_t)b * M + s] = nxt;
      newxyz[((size_t)b * M + s) * 3 + 0] = nx;
      newxyz[((size_t)b * M + s) * 3 + 1] = ny;
      newxyz[((size_t)b * M + s) * 3 + 2] = nz;
    }
    for (int j = 0; j < cnt; ++j) {
      float dx = px[j] - nx, dy = py[j] - ny, dz = pz[j] - nz;
      dm[j] = fminf(dm[j], dx * dx + dy * dy + dz * dz);
    }
    __syncthreads();
  }
}

// ------------------- ball query: one wave per centroid, first 16 within radius -------------------
__global__ void ballquery_kernel(const float* __restrict__ newxyz, const float* __restrict__ xyz,
                                 int M, int Nsrc, float r2, int* __restrict__ gi) {
  int wid = (int)((blockIdx.x * blockDim.x + threadIdx.x) >> 5);
  if (wid >= BB * M) return;
  int lane = lane_id();
  int b = wid / M, m = wid % M;
  const float* q = newxyz + ((size_t)b * M + m) * 3;
  float qx = q[0], qy = q[1], qz = q[2];
  const float* X = xyz + (size_t)b * Nsrc * 3;
  int* out = gi + ((size_t)b * M + m) * 16;
  int cnt = 0, first = -1;
  for (int base = 0; base < Nsrc && cnt < 16; base += 32) {
    int i = base + lane;
    bool pred = false;
    if (i < Nsrc) {
      float dx = X[i * 3] - qx, dy = X[i * 3 + 1] - qy, dz = X[i * 3 + 2] - qz;
      pred = (dx * dx + dy * dy + dz * dz) <= r2;
    }
    unsigned mask = __builtin_amdgcn_ballot_w32(pred);
    while (mask && cnt < 16) {
      int j = __builtin_ctz(mask);
      mask &= mask - 1;
      if (first < 0) first = base + j;
      if (lane == 0) out[cnt] = base + j;
      ++cnt;
    }
  }
  if (lane == 0) {
    if (first < 0) first = 0;
    for (int c = cnt; c < 16; ++c) out[c] = first;
  }
}

// ------------------- kNN (k=8): one thread per query, insertion top-8 -------------------
__global__ void knn_kernel(const float* __restrict__ q, const float* __restrict__ ref,
                           int Nq, int Ns, int* __restrict__ gi) {
  int t = blockIdx.x * blockDim.x + threadIdx.x;
  if (t >= BB * Nq) return;
  int b = t / Nq, n = t % Nq;
  const float* Q = q + ((size_t)b * Nq + n) * 3;
  float qx = Q[0], qy = Q[1], qz = Q[2];
  const float* R = ref + (size_t)b * Ns * 3;
  float bd[8]; int bi[8];
#pragma unroll
  for (int j = 0; j < 8; ++j) { bd[j] = 1e30f; bi[j] = 0; }
  for (int i = 0; i < Ns; ++i) {
    float dx = R[i * 3] - qx, dy = R[i * 3 + 1] - qy, dz = R[i * 3 + 2] - qz;
    float d = dx * dx + dy * dy + dz * dz;
    if (d < bd[7]) {
      int p = 7;
      while (p > 0 && bd[p - 1] > d) { bd[p] = bd[p - 1]; bi[p] = bi[p - 1]; --p; }
      bd[p] = d; bi[p] = i;
    }
  }
  int* o = gi + ((size_t)b * Nq + n) * 8;
#pragma unroll
  for (int j = 0; j < 8; ++j) o[j] = bi[j];
}

// ------------------- SetConv: gather -> MLP(WMMA) -> MLP(WMMA) -> maxpool(16) -------------------
// block = 64 threads (2 waves), 2 groups x 16 samples = 32 rows
template <int CF, int C1, int C2, int K1T>
__global__ void setconv_kernel(const float* __restrict__ xyz, const float* __restrict__ feat,
                               const float* __restrict__ newxyz, const int* __restrict__ gi,
                               const _Float16* __restrict__ pw1, const float* __restrict__ b1,
                               const _Float16* __restrict__ pw2, const float* __restrict__ b2,
                               float* __restrict__ fout, int Nsrc, int M) {
  constexpr int CIN1 = 3 + CF;
  constexpr int K1P = K1T * 32;
  constexpr int ROWS = 32;
  __shared__ _Float16 h1[ROWS * K1P];
  __shared__ _Float16 h2[ROWS * C1];
  __shared__ _Float16 ho[ROWS * C2];
  int b = blockIdx.y;
  int m0 = blockIdx.x * 2;
  int tid = threadIdx.x;
  if (tid < ROWS) {
    int g = tid >> 4, kk = tid & 15;
    int m = m0 + g;
    int gidx = gi[((size_t)b * M + m) * 16 + kk];
    const float* p = xyz + ((size_t)b * Nsrc + gidx) * 3;
    const float* nx = newxyz + ((size_t)b * M + m) * 3;
    _Float16* row = h1 + (size_t)tid * K1P;
    row[0] = (_Float16)(p[0] - nx[0]);
    row[1] = (_Float16)(p[1] - nx[1]);
    row[2] = (_Float16)(p[2] - nx[2]);
    const float* f = feat + ((size_t)b * Nsrc + gidx) * CF;
    for (int c = 0; c < CF; ++c) row[3 + c] = (_Float16)f[c];
    for (int c = CIN1; c < K1P; ++c) row[c] = (_Float16)0.f;
  }
  __syncthreads();
  lds_gemm_relu<ROWS, K1T, C1>(h1, K1P, pw1, b1, h2, C1, 2);
  __syncthreads();
  lds_gemm_relu<ROWS, C1 / 32, C2>(h2, C1, pw2, b2, ho, C2, 2);
  __syncthreads();
  for (int i = tid; i < 2 * C2; i += 64) {
    int g = i / C2, c = i % C2;
    float mx = -1e30f;
    for (int kk = 0; kk < 16; ++kk)
      mx = fmaxf(mx, (float)ho[(size_t)(g * 16 + kk) * C2 + c]);
    fout[((size_t)b * M + m0 + g) * C2 + c] = mx;
  }
}

// ------------------- SetUpConv: gather -> MLP(WMMA) -> maxpool(8) -> concat -> MLP(WMMA) ---------
// block = 64 threads (2 waves), GD=4 dense points x 8 neighbors = 32 rows for GEMM1
template <int CS, int C1, int CD, int C2, int K1T>
__global__ void upconv_kernel(const float* __restrict__ xyz_s, const float* __restrict__ feat_s,
                              const float* __restrict__ xyz_d, const float* __restrict__ feat_d,
                              const int* __restrict__ gi,
                              const _Float16* __restrict__ pw1, const float* __restrict__ b1,
                              const _Float16* __restrict__ pw2, const float* __restrict__ b2,
                              float* __restrict__ uout, int Ns, int Nd) {
  constexpr int CIN1 = 3 + CS;
  constexpr int K1P = K1T * 32;
  constexpr int GD = 4;
  constexpr int ROWS = GD * 8; // 32
  constexpr int K2 = C1 + CD;  // multiple of 32
  __shared__ _Float16 h1[ROWS * K1P];
  __shared__ _Float16 mid[ROWS * C1];
  __shared__ _Float16 h2[16 * K2];
  __shared__ _Float16 ho[16 * C2];
  int b = blockIdx.y;
  int d0 = blockIdx.x * GD;
  int tid = threadIdx.x;
  if (tid < ROWS) {
    int dl = tid >> 3, kk = tid & 7;
    int d = d0 + dl;
    int gidx = gi[((size_t)b * Nd + d) * 8 + kk];
    const float* p = xyz_s + ((size_t)b * Ns + gidx) * 3;
    const float* q = xyz_d + ((size_t)b * Nd + d) * 3;
    _Float16* row = h1 + (size_t)tid * K1P;
    row[0] = (_Float16)(p[0] - q[0]);
    row[1] = (_Float16)(p[1] - q[1]);
    row[2] = (_Float16)(p[2] - q[2]);
    const float* f = feat_s + ((size_t)b * Ns + gidx) * CS;
    for (int c = 0; c < CS; ++c) row[3 + c] = (_Float16)f[c];
    for (int c = CIN1; c < K1P; ++c) row[c] = (_Float16)0.f;
  }
  __syncthreads();
  lds_gemm_relu<ROWS, K1T, C1>(h1, K1P, pw1, b1, mid, C1, 2);
  __syncthreads();
  for (int i = tid; i < 16 * K2; i += 64) {
    int p = i / K2, c = i % K2;
    float v = 0.f;
    if (p < GD) {
      if (c < C1) {
        float mx = -1e30f;
        for (int kk = 0; kk < 8; ++kk)
          mx = fmaxf(mx, (float)mid[(size_t)(p * 8 + kk) * C1 + c]);
        v = mx;
      } else {
        v = feat_d[((size_t)b * Nd + d0 + p) * CD + (c - C1)];
      }
    }
    h2[i] = (_Float16)v;
  }
  __syncthreads();
  lds_gemm_relu<16, K2 / 32, C2>(h2, K2, pw2, b2, ho, C2, 2);
  __syncthreads();
  for (int i = tid; i < GD * C2; i += 64) {
    int p = i / C2, c = i % C2;
    uout[((size_t)b * Nd + d0 + p) * C2 + c] = (float)ho[(size_t)p * C2 + c];
  }
}

// ------------------- output writers -------------------
__global__ void transpose_out_kernel(const float* __restrict__ in, float* __restrict__ out,
                                     int Nn, int C) {
  int t = blockIdx.x * blockDim.x + threadIdx.x;
  if (t >= BB * Nn * C) return;
  int b = t / (Nn * C);
  int rem = t % (Nn * C);
  int n = rem / C, c = rem % C;
  out[((size_t)b * C + c) * Nn + n] = in[((size_t)b * Nn + n) * C + c];
}

__global__ void idx_out_kernel(const int* __restrict__ in, float* __restrict__ out, int n) {
  int t = blockIdx.x * blockDim.x + threadIdx.x;
  if (t < n) out[t] = (float)in[t];
}

// ------------------- host launcher -------------------
extern "C" void kernel_launch(void* const* d_in, const int* in_sizes, int n_in,
                              void* d_out, int out_size, void* d_ws, size_t ws_size,
                              hipStream_t stream) {
  (void)in_sizes; (void)n_in; (void)out_size; (void)ws_size;
  const float* pc   = (const float*)d_in[0];
  const float* feat = (const float*)d_in[1];
  auto W = [&](int i) { return (const float*)d_in[i]; };
  float* out = (float*)d_out;
  char* ws = (char*)d_ws;
  size_t off = 0;
  auto allocf = [&](size_t n) { float* p = (float*)(ws + off); off += ((n * 4 + 255) / 256) * 256; return p; };
  auto alloci = [&](size_t n) { int* p = (int*)(ws + off); off += ((n * 4 + 255) / 256) * 256; return p; };
  auto alloch = [&](size_t n) { _Float16* p = (_Float16*)(ws + off); off += ((n * 2 + 255) / 256) * 256; return p; };

  float* xyz0 = allocf(BB * 8192 * 3);
  float* f0   = allocf((size_t)BB * 8192 * 32);
  float* xyz1 = allocf(BB * 2048 * 3);
  float* xyz2 = allocf(BB * 512 * 3);
  float* xyz3 = allocf(BB * 128 * 3);
  float* xyz4 = allocf(BB * 64 * 3);
  int* idx1 = alloci(BB * 2048);
  int* idx2 = alloci(BB * 512);
  int* idx3 = alloci(BB * 128);
  int* idx4 = alloci(BB * 64);
  int* gi1 = alloci((size_t)BB * 2048 * 16);
  int* gi2 = alloci((size_t)BB * 512 * 16);
  int* gi3 = alloci((size_t)BB * 128 * 16);
  int* gi4 = alloci((size_t)BB * 64 * 16);
  float* f1 = allocf((size_t)BB * 2048 * 64);
  float* f2 = allocf((size_t)BB * 512 * 128);
  float* f3 = allocf((size_t)BB * 128 * 192);
  float* f4 = allocf((size_t)BB * 64 * 192);
  int* kn3 = alloci((size_t)BB * 128 * 8);
  int* kn2 = alloci((size_t)BB * 512 * 8);
  int* kn1 = alloci((size_t)BB * 2048 * 8);
  int* kn0 = alloci((size_t)BB * 8192 * 8);
  float* u3 = allocf((size_t)BB * 128 * 192);
  float* u2 = allocf((size_t)BB * 512 * 128);
  float* u1 = allocf((size_t)BB * 2048 * 64);
  float* u0 = allocf((size_t)BB * 8192 * 32);

  // packed f16 weights (zero-padded K)
  _Float16* p_sc1w1 = alloch(32 * 64);
  _Float16* p_sc1w2 = alloch(64 * 32);
  _Float16* p_sc2w1 = alloch(64 * 96);
  _Float16* p_sc2w2 = alloch(128 * 64);
  _Float16* p_sc3w1 = alloch(128 * 160);
  _Float16* p_sc3w2 = alloch(192 * 128);
  _Float16* p_sc4w1 = alloch(192 * 224);
  _Float16* p_sc4w2 = alloch(192 * 192);
  _Float16* p_up4w1 = alloch(192 * 224);
  _Float16* p_up4w2 = alloch(192 * 384);
  _Float16* p_up3w1 = alloch(128 * 224);
  _Float16* p_up3w2 = alloch(128 * 256);
  _Float16* p_up2w1 = alloch(64 * 160);
  _Float16* p_up2w2 = alloch(64 * 128);
  _Float16* p_up1w1 = alloch(32 * 96);
  _Float16* p_up1w2 = alloch(32 * 64);

  auto pack = [&](const float* w, _Float16* o, int Cout, int Cin, int KP) {
    int tot = Cout * KP;
    pack_w_kernel<<<(tot + 255) / 256, 256, 0, stream>>>(w, o, Cout, Cin, KP);
  };
  pack(W(4),  p_sc1w1, 32, 35, 64);    pack(W(6),  p_sc1w2, 64, 32, 32);
  pack(W(8),  p_sc2w1, 64, 67, 96);    pack(W(10), p_sc2w2, 128, 64, 64);
  pack(W(12), p_sc3w1, 128, 131, 160); pack(W(14), p_sc3w2, 192, 128, 128);
  pack(W(16), p_sc4w1, 192, 195, 224); pack(W(18), p_sc4w2, 192, 192, 192);
  pack(W(20), p_up4w1, 192, 195, 224); pack(W(22), p_up4w2, 192, 384, 384);
  pack(W(24), p_up3w1, 128, 195, 224); pack(W(26), p_up3w2, 128, 256, 256);
  pack(W(28), p_up2w1, 64, 131, 160);  pack(W(30), p_up2w2, 64, 128, 128);
  pack(W(32), p_up1w1, 32, 67, 96);    pack(W(34), p_up1w2, 32, 64, 64);

  // init: transpose + down0 MLP
  init_kernel<<<(BB * NN + 255) / 256, 256, 0, stream>>>(pc, feat, W(2), W(3), xyz0, f0);

  // ---- encoder down path ----
  fps_kernel<<<BB, 1024, 0, stream>>>(xyz0, 8192, 2048, idx1, xyz1);
  ballquery_kernel<<<(BB * 2048 * 32 + 255) / 256, 256, 0, stream>>>(xyz1, xyz0, 2048, 8192, 0.25f, gi1);
  setconv_kernel<32, 32, 64, 2><<<dim3(1024, BB), 64, 0, stream>>>(
      xyz0, f0, xyz1, gi1, p_sc1w1, W(5), p_sc1w2, W(7), f1, 8192, 2048);

  fps_kernel<<<BB, 1024, 0, stream>>>(xyz1, 2048, 512, idx2, xyz2);
  ballquery_kernel<<<(BB * 512 * 32 + 255) / 256, 256, 0, stream>>>(xyz2, xyz1, 512, 2048, 1.0f, gi2);
  setconv_kernel<64, 64, 128, 3><<<dim3(256, BB), 64, 0, stream>>>(
      xyz1, f1, xyz2, gi2, p_sc2w1, W(9), p_sc2w2, W(11), f2, 2048, 512);

  fps_kernel<<<BB, 1024, 0, stream>>>(xyz2, 512, 128, idx3, xyz3);
  ballquery_kernel<<<(BB * 128 * 32 + 255) / 256, 256, 0, stream>>>(xyz3, xyz2, 128, 512, 4.0f, gi3);
  setconv_kernel<128, 128, 192, 5><<<dim3(64, BB), 64, 0, stream>>>(
      xyz2, f2, xyz3, gi3, p_sc3w1, W(13), p_sc3w2, W(15), f3, 512, 128);

  fps_kernel<<<BB, 1024, 0, stream>>>(xyz3, 128, 64, idx4, xyz4);
  ballquery_kernel<<<(BB * 64 * 32 + 255) / 256, 256, 0, stream>>>(xyz4, xyz3, 64, 128, 16.0f, gi4);
  setconv_kernel<192, 192, 192, 7><<<dim3(32, BB), 64, 0, stream>>>(
      xyz3, f3, xyz4, gi4, p_sc4w1, W(17), p_sc4w2, W(19), f4, 128, 64);

  // ---- decoder up path ----
  knn_kernel<<<(BB * 128 + 255) / 256, 256, 0, stream>>>(xyz3, xyz4, 128, 64, kn3);
  upconv_kernel<192, 192, 192, 192, 7><<<dim3(32, BB), 64, 0, stream>>>(
      xyz4, f4, xyz3, f3, kn3, p_up4w1, W(21), p_up4w2, W(23), u3, 64, 128);

  knn_kernel<<<(BB * 512 + 255) / 256, 256, 0, stream>>>(xyz2, xyz3, 512, 128, kn2);
  upconv_kernel<192, 128, 128, 128, 7><<<dim3(128, BB), 64, 0, stream>>>(
      xyz3, u3, xyz2, f2, kn2, p_up3w1, W(25), p_up3w2, W(27), u2, 128, 512);

  knn_kernel<<<(BB * 2048 + 255) / 256, 256, 0, stream>>>(xyz1, xyz2, 2048, 512, kn1);
  upconv_kernel<128, 64, 64, 64, 5><<<dim3(512, BB), 64, 0, stream>>>(
      xyz2, u2, xyz1, f1, kn1, p_up2w1, W(29), p_up2w2, W(31), u1, 512, 2048);

  knn_kernel<<<(BB * 8192 + 255) / 256, 256, 0, stream>>>(xyz0, xyz1, 8192, 2048, kn0);
  upconv_kernel<64, 32, 32, 32, 3><<<dim3(2048, BB), 64, 0, stream>>>(
      xyz1, u1, xyz0, f0, kn0, p_up1w1, W(33), p_up1w2, W(35), u0, 2048, 8192);

  // ---- outputs: (x1,x2,x3, i1,i2,i3, u0,u1,u2,u3) flattened ----
  auto tlaunch = [&](const float* in, float* o, int Nn, int C) {
    int tot = BB * Nn * C;
    transpose_out_kernel<<<(tot + 255) / 256, 256, 0, stream>>>(in, o, Nn, C);
  };
  size_t p = 0;
  tlaunch(xyz1, out + p, 2048, 3); p += (size_t)BB * 3 * 2048;
  tlaunch(xyz2, out + p, 512, 3);  p += (size_t)BB * 3 * 512;
  tlaunch(xyz3, out + p, 128, 3);  p += (size_t)BB * 3 * 128;
  idx_out_kernel<<<(BB * 2048 + 255) / 256, 256, 0, stream>>>(idx1, out + p, BB * 2048); p += BB * 2048;
  idx_out_kernel<<<(BB * 512 + 255) / 256, 256, 0, stream>>>(idx2, out + p, BB * 512);   p += BB * 512;
  idx_out_kernel<<<(BB * 128 + 255) / 256, 256, 0, stream>>>(idx3, out + p, BB * 128);   p += BB * 128;
  tlaunch(u0, out + p, 8192, 32);  p += (size_t)BB * 32 * 8192;
  tlaunch(u1, out + p, 2048, 64);  p += (size_t)BB * 64 * 2048;
  tlaunch(u2, out + p, 512, 128);  p += (size_t)BB * 128 * 512;
  tlaunch(u3, out + p, 128, 192);  p += (size_t)BB * 192 * 128;
}
